// BWRPE_ater_19868518711491
// MI455X (gfx1250) — matrix operand order repaired and checked by reading, not compiled, not run on hardware
//
#include <hip/hip_runtime.h>
#include <hip/hip_bf16.h>
#include <math.h>

// ---------------- problem constants ----------------
constexpr int S_   = 5;
constexpr int B_   = 2;
constexpr int N_   = 1024;
constexpr int C_   = 768;
constexpr int H_   = 12;
constexpr int HID_ = 3072;
constexpr int QKVW = 3 * C_;            // 2304
constexpr int T_   = S_ * B_ * N_;      // 10240 rows
constexpr float SCALE_ = 0.125f;        // (C/H)^-0.5 = 64^-0.5

typedef __bf16 bf16;
typedef __attribute__((ext_vector_type(16))) __bf16        v16bf;
typedef __attribute__((ext_vector_type(8)))  float         v8f;
typedef __attribute__((ext_vector_type(4)))  unsigned int  v4u;

// ---------------- vectorized WMMA fragment loaders (ISA 7.12.2 layouts) ----------------
// Fragment element order:
//  A (16x32): lane {m=lane&15, half=lane>>4}; elems 0..7 = A[m][half*8+0..7],
//             elems 8..15 = A[m][16+half*8+0..7]  -> two contiguous 16B runs.
//  B (32x16): lane {n=lane&15, half=lane>>4}; elems 0..7 = B[half*8+0..7][n],
//             elems 8..15 = B[16+half*8+0..7][n] -> with n-major storage two 16B runs.
__device__ __forceinline__ v16bf frag_from2(const bf16* p0, const bf16* p1) {
  union { v4u u[2]; v16bf v; } r;
  r.u[0] = *(const v4u*)p0;
  r.u[1] = *(const v4u*)p1;
  return r.v;
}
__device__ __forceinline__ v16bf load_a_frag_v(const bf16* base, int strideM) {
  int lane = threadIdx.x & 31;
  const bf16* r = base + (lane & 15) * strideM + (lane >> 4) * 8;
  return frag_from2(r, r + 16);
}
__device__ __forceinline__ v16bf load_b_frag_v(const bf16* base, int strideN) {
  int lane = threadIdx.x & 31;
  const bf16* r = base + (lane & 15) * strideN + (lane >> 4) * 8;
  return frag_from2(r, r + 16);
}
__device__ __forceinline__ v8f wmma_bf16(v16bf a, v16bf b, v8f c) {
  return __builtin_amdgcn_wmma_f32_16x16x32_bf16(false, a, false, b, (short)0, c, false, false);
}

// ---------------- LayerNorm -> bf16 ----------------
__global__ __launch_bounds__(256)
void ln_kernel(const float* __restrict__ x, const float* __restrict__ g,
               const float* __restrict__ b, bf16* __restrict__ out) {
  int row = blockIdx.x, tid = threadIdx.x;
  const float* xr = x + (size_t)row * C_;
  float v[3], s = 0.f, ss = 0.f;
#pragma unroll
  for (int i = 0; i < 3; ++i) { v[i] = xr[tid + i * 256]; s += v[i]; ss += v[i] * v[i]; }
  __shared__ float r1[256], r2[256];
  r1[tid] = s; r2[tid] = ss;
  __syncthreads();
  for (int off = 128; off > 0; off >>= 1) {
    if (tid < off) { r1[tid] += r1[tid + off]; r2[tid] += r2[tid + off]; }
    __syncthreads();
  }
  float mu = r1[0] * (1.f / C_);
  float var = r2[0] * (1.f / C_) - mu * mu;
  float rstd = rsqrtf(var + 1e-5f);
  bf16* orow = out + (size_t)row * C_;
#pragma unroll
  for (int i = 0; i < 3; ++i) {
    int c = tid + i * 256;
    orow[c] = (bf16)((v[i] - mu) * rstd * g[c] + b[c]);
  }
}

// ---------------- adapter down(Cx8) + mid(8x8) -> h2[T,8] ----------------
__global__ __launch_bounds__(64)
void adapter_kernel(const bf16* __restrict__ act, const float* __restrict__ dw,
                    const float* __restrict__ db, const float* __restrict__ mw,
                    const float* __restrict__ mb, float* __restrict__ h2) {
  int row = blockIdx.x, tid = threadIdx.x;
  int j = tid >> 3, p = tid & 7;
  const bf16* ar = act + (size_t)row * C_;
  float s = 0.f;
  for (int c = p; c < C_; c += 8) s += (float)ar[c] * dw[c * 8 + j];
  __shared__ float part[8][8];
  __shared__ float hrow[8];
  part[j][p] = s;
  __syncthreads();
  if (tid < 8) {
    float h = db[tid];
#pragma unroll
    for (int q = 0; q < 8; ++q) h += part[tid][q];
    hrow[tid] = h;
  }
  __syncthreads();
  if (tid < 8) {
    float v = mb[tid];
#pragma unroll
    for (int i = 0; i < 8; ++i) v += hrow[i] * mw[i * 8 + tid];
    h2[(size_t)row * 8 + tid] = v;
  }
}

// ---------------- WMMA GEMM: out[M,N] = A(bf16)[M,K] @ W(f32->bf16)[K,N] + bias ----------------
// emode: 0 = f32 out, 1 = bf16 out, 2 = bf16 out with exact GELU,
//        3 = QKV: cols [0,1536) -> bf16 out; cols [1536,2304) -> vT[sb, c-1536, key] (transposed V)
__global__ __launch_bounds__(256)
void gemm_bf16_kernel(const bf16* __restrict__ A, const float* __restrict__ W,
                      const float* __restrict__ bias, void* __restrict__ outp,
                      bf16* __restrict__ vT,
                      int M, int N, int K, int emode) {
  constexpr int TM = 128, TN = 64, TK = 32, LDB = TK + 8;
  __shared__ bf16 sA[TM][TK];     // 8 KB, row-major
  __shared__ bf16 sBt[TN][LDB];   // 5 KB, n-major (fragment-friendly)
  int tid = threadIdx.x;
  int bm = blockIdx.y * TM, bn = blockIdx.x * TN;
  int wid = tid >> 5, lane = tid & 31;
  int wm = (wid & 3) * 32, wn = (wid >> 2) * 32;
  v8f acc[2][2] = {};
  // A staging indices: 16 contiguous bf16 (32B) per thread
  int arow = (tid * 16) >> 5, acol = (tid * 16) & 31;
  for (int k0 = 0; k0 < K; k0 += TK) {
    {
      const bf16* g = A + (size_t)(bm + arow) * K + k0 + acol;
      if (k0 + TK < K) __builtin_prefetch(g + TK, 0, 1);
      *(v4u*)&sA[arow][acol]     = *(const v4u*)g;
      *(v4u*)&sA[arow][acol + 8] = *(const v4u*)(g + 8);
    }
    {   // W: coalesced f32 reads, transposed bf16 LDS writes
#pragma unroll
      for (int i = 0; i < 8; ++i) {
        int e = tid + 256 * i;
        int n = e & 63, k = e >> 6;
        sBt[n][k] = (bf16)W[(size_t)(k0 + k) * N + bn + n];
      }
    }
    __syncthreads();
    v16bf aF[2], bF[2];
#pragma unroll
    for (int mi = 0; mi < 2; ++mi) aF[mi] = load_a_frag_v(&sA[wm + mi * 16][0], TK);
#pragma unroll
    for (int ni = 0; ni < 2; ++ni) bF[ni] = load_b_frag_v(&sBt[wn + ni * 16][0], LDB);
#pragma unroll
    for (int mi = 0; mi < 2; ++mi)
#pragma unroll
      for (int ni = 0; ni < 2; ++ni)
        acc[mi][ni] = wmma_bf16(aF[mi], bF[ni], acc[mi][ni]);
    __syncthreads();
  }
  int lm = lane >> 4, ln = lane & 15;
#pragma unroll
  for (int mi = 0; mi < 2; ++mi)
#pragma unroll
    for (int ni = 0; ni < 2; ++ni)
#pragma unroll
      for (int r = 0; r < 8; ++r) {
        int gm = bm + wm + mi * 16 + r + 8 * lm;
        int gn = bn + wn + ni * 16 + ln;
        float v = acc[mi][ni][r] + bias[gn];
        if (emode == 2) v = 0.5f * v * (1.f + erff(v * 0.70710678118f));
        if (emode == 0) {
          ((float*)outp)[(size_t)gm * N + gn] = v;
        } else if (emode == 3 && gn >= 2 * C_) {
          int sb = gm >> 10, key = gm & (N_ - 1);
          vT[((size_t)sb * C_ + (gn - 2 * C_)) * N_ + key] = (bf16)v;
        } else {
          ((bf16*)outp)[(size_t)gm * N + gn] = (bf16)v;
        }
      }
}

// ---------------- attention: 1 wave = (sb, h, 16-row q block), flash softmax ----------------
// Processes 32 real keys per iteration: 2 score tiles -> full 16x32 P -> one PV pass.
__global__ __launch_bounds__(32)
void attn_kernel(const bf16* __restrict__ qkv, const bf16* __restrict__ vT,
                 bf16* __restrict__ out) {
  int qb = blockIdx.x & 63;            // N/16 = 64 q tiles
  int hh = blockIdx.x >> 6;
  int h  = hh % H_;
  int sb = hh / H_;
  int lane = threadIdx.x & 31;
  int half = lane >> 4, ln16 = lane & 15;

  __shared__ bf16 pbuf[16][32];

  const bf16* qbase = qkv + (size_t)(sb * N_ + qb * 16) * QKVW + h * 64;
  v16bf qa0 = load_a_frag_v(qbase,      QKVW);
  v16bf qa1 = load_a_frag_v(qbase + 32, QKVW);

  float mrow[8], lrow[8];
  v8f oacc[4] = {};
#pragma unroll
  for (int r = 0; r < 8; ++r) { mrow[r] = -1e30f; lrow[r] = 0.f; }

  const bf16* vtb = vT + (size_t)(sb * C_ + h * 64) * N_;

  for (int kc = 0; kc < N_ / 32; ++kc) {
    int key0 = kc * 32;
    const bf16* kb0 = qkv + (size_t)(sb * N_ + key0) * QKVW + C_ + h * 64;
    const bf16* kb1 = kb0 + (size_t)16 * QKVW;
    // S tiles: Q(16x64) @ K^T(64x16); K is dim-contiguous -> direct b128 frags
    v8f s0 = {}, s1 = {};
    s0 = wmma_bf16(qa0, load_b_frag_v(kb0,      QKVW), s0);
    s0 = wmma_bf16(qa1, load_b_frag_v(kb0 + 32, QKVW), s0);
    s1 = wmma_bf16(qa0, load_b_frag_v(kb1,      QKVW), s1);
    s1 = wmma_bf16(qa1, load_b_frag_v(kb1 + 32, QKVW), s1);
#pragma unroll
    for (int r = 0; r < 8; ++r) {
      float sv0 = s0[r] * SCALE_;
      float sv1 = s1[r] * SCALE_;
      float mx = fmaxf(sv0, sv1);
#pragma unroll
      for (int off = 1; off < 16; off <<= 1) mx = fmaxf(mx, __shfl_xor(mx, off, 32));
      float mnew = fmaxf(mrow[r], mx);
      float corr = __expf(mrow[r] - mnew);
      float p0 = __expf(sv0 - mnew);
      float p1 = __expf(sv1 - mnew);
      float rs = p0 + p1;
#pragma unroll
      for (int off = 1; off < 16; off <<= 1) rs += __shfl_xor(rs, off, 32);
      lrow[r] = lrow[r] * corr + rs;
      mrow[r] = mnew;
#pragma unroll
      for (int nt = 0; nt < 4; ++nt) oacc[nt][r] = oacc[nt][r] * corr;
      pbuf[r + 8 * half][ln16]      = (bf16)p0;
      pbuf[r + 8 * half][16 + ln16] = (bf16)p1;
    }
    __syncthreads();
    v16bf pa = load_a_frag_v(&pbuf[0][0], 32);
#pragma unroll
    for (int nt = 0; nt < 4; ++nt)   // V transposed: key-contiguous -> b128 frags
      oacc[nt] = wmma_bf16(pa, load_b_frag_v(vtb + (size_t)(nt * 16) * N_ + key0, N_), oacc[nt]);
    __syncthreads();
  }
#pragma unroll
  for (int r = 0; r < 8; ++r) {
    float inv = 1.f / lrow[r];
    int m = r + 8 * half;
    size_t t = (size_t)(sb * N_ + qb * 16 + m);
#pragma unroll
    for (int nt = 0; nt < 4; ++nt)
      out[t * C_ + h * 64 + nt * 16 + ln16] = (bf16)(oacc[nt][r] * inv);
  }
}

// ---------------- residual combine with cross-stream adapter-up ----------------
__global__ __launch_bounds__(256)
void combine_kernel(const float* x0, const float* x1, const float* x2,
                    const float* x3, const float* x4,
                    const float* __restrict__ gem, const float* __restrict__ h2,
                    const float* __restrict__ uw, const float* __restrict__ ub,
                    float* __restrict__ outy) {
  long idx = (long)blockIdx.x * blockDim.x + threadIdx.x;
  if (idx >= (long)T_ * C_) return;
  int c = (int)(idx % C_);
  int t = (int)(idx / C_);
  int s = t / (B_ * N_);
  int rem = t % (B_ * N_);
  const float* xs = (s == 0) ? x0 : (s == 1) ? x1 : (s == 2) ? x2 : (s == 3) ? x3 : x4;
  float v = xs[(size_t)rem * C_ + c] + gem[idx];
#pragma unroll
  for (int d = -1; d <= 1; d += 2) {
    int s2 = s + d;
    if (0 <= s2 && s2 < S_) {
      const float* hh = h2 + ((size_t)(s2 * B_ * N_ + rem)) * 8;
      float a = ub[c];
#pragma unroll
      for (int i = 0; i < 8; ++i) a += hh[i] * uw[i * C_ + c];
      v += a;
    }
  }
  outy[idx] = v;
}

// ---------------- host launcher ----------------
extern "C" void kernel_launch(void* const* d_in, const int* in_sizes, int n_in,
                              void* d_out, int out_size, void* d_ws, size_t ws_size,
                              hipStream_t stream) {
  const float* x[5];
  for (int i = 0; i < 5; ++i) x[i] = (const float*)d_in[i];
  const float* ln1_g = (const float*)d_in[5];
  const float* ln1_b = (const float*)d_in[6];
  const float* ln2_g = (const float*)d_in[7];
  const float* ln2_b = (const float*)d_in[8];
  const float* qkv_w = (const float*)d_in[9];
  const float* qkv_b = (const float*)d_in[10];
  const float* proj_w = (const float*)d_in[11];
  const float* proj_b = (const float*)d_in[12];
  const float* fc1_w = (const float*)d_in[13];
  const float* fc1_b = (const float*)d_in[14];
  const float* fc2_w = (const float*)d_in[15];
  const float* fc2_b = (const float*)d_in[16];
  const float* at_dw = (const float*)d_in[17];
  const float* at_db = (const float*)d_in[18];
  const float* at_mw = (const float*)d_in[19];
  const float* at_mb = (const float*)d_in[20];
  const float* at_uw = (const float*)d_in[21];
  const float* at_ub = (const float*)d_in[22];
  const float* a2_dw = (const float*)d_in[23];
  const float* a2_db = (const float*)d_in[24];
  const float* a2_mw = (const float*)d_in[25];
  const float* a2_mb = (const float*)d_in[26];
  const float* a2_uw = (const float*)d_in[27];
  const float* a2_ub = (const float*)d_in[28];

  char* ws = (char*)d_ws;
  // persistent region
  const size_t oYS  = 0;                                    // f32 [T,C]
  const size_t oH2A = oYS  + (size_t)T_ * C_ * 4;           // f32 [T,8]
  const size_t oH2B = oH2A + (size_t)T_ * 8 * 4;
  const size_t oR   = oH2B + (size_t)T_ * 8 * 4;            // overlaid region
  // phase 1 overlay
  const size_t oN1   = oR;                                  // bf16 [T,C]
  const size_t oQKV  = oN1   + (size_t)T_ * C_ * 2;         // bf16 [T,2304] (V part unused)
  const size_t oATT  = oQKV  + (size_t)T_ * QKVW * 2;       // bf16 [T,C]
  const size_t oPROJ = oATT  + (size_t)T_ * C_ * 2;         // f32  [T,C]
  const size_t oVT   = oPROJ + (size_t)T_ * C_ * 4;         // bf16 [S*B, C, N] transposed V
  // phase 2 overlay (phase-1 buffers dead by then)
  const size_t oLN1Y = oR;                                  // bf16 [T,C]
  const size_t oLN2Y = oLN1Y + (size_t)T_ * C_ * 2;         // bf16 [T,C]
  const size_t oGELU = oLN2Y + (size_t)T_ * C_ * 2;         // bf16 [T,HID]
  const size_t oMLP  = oGELU + (size_t)T_ * HID_ * 2;       // f32  [T,C]

  // 1) LN1 on each input stream -> n1 (bf16)
  for (int s = 0; s < S_; ++s)
    ln_kernel<<<B_ * N_, 256, 0, stream>>>(x[s], ln1_g, ln1_b,
        (bf16*)(ws + oN1) + (size_t)s * B_ * N_ * C_);
  // 2) adapter-1 down+mid
  adapter_kernel<<<T_, 64, 0, stream>>>((const bf16*)(ws + oN1), at_dw, at_db,
                                        at_mw, at_mb, (float*)(ws + oH2A));
  // 3) QKV GEMM: Q,K -> [T,2304] bf16; V -> transposed vT
  gemm_bf16_kernel<<<dim3(QKVW / 64, T_ / 128), 256, 0, stream>>>(
      (const bf16*)(ws + oN1), qkv_w, qkv_b, ws + oQKV, (bf16*)(ws + oVT),
      T_, QKVW, C_, 3);
  // 4) attention
  attn_kernel<<<(S_ * B_ * H_) * (N_ / 16), 32, 0, stream>>>(
      (const bf16*)(ws + oQKV), (const bf16*)(ws + oVT), (bf16*)(ws + oATT));
  // 5) proj GEMM (f32 out)
  gemm_bf16_kernel<<<dim3(C_ / 64, T_ / 128), 256, 0, stream>>>(
      (const bf16*)(ws + oATT), proj_w, proj_b, ws + oPROJ, nullptr, T_, C_, C_, 0);
  // 6) ys = x + proj + cross a1
  long totalTC = (long)T_ * C_;
  combine_kernel<<<(unsigned)((totalTC + 255) / 256), 256, 0, stream>>>(
      x[0], x[1], x[2], x[3], x[4], (const float*)(ws + oPROJ),
      (const float*)(ws + oH2A), at_uw, at_ub, (float*)(ws + oYS));
  // 7) LN(ys) with ln1 params (adapter-2 input) and ln2 params (MLP input)
  ln_kernel<<<T_, 256, 0, stream>>>((const float*)(ws + oYS), ln1_g, ln1_b, (bf16*)(ws + oLN1Y));
  ln_kernel<<<T_, 256, 0, stream>>>((const float*)(ws + oYS), ln2_g, ln2_b, (bf16*)(ws + oLN2Y));
  // 8) adapter-2 down+mid
  adapter_kernel<<<T_, 64, 0, stream>>>((const bf16*)(ws + oLN1Y), a2_dw, a2_db,
                                        a2_mw, a2_mb, (float*)(ws + oH2B));
  // 9) FC1 + exact GELU (bf16 out)
  gemm_bf16_kernel<<<dim3(HID_ / 64, T_ / 128), 256, 0, stream>>>(
      (const bf16*)(ws + oLN2Y), fc1_w, fc1_b, ws + oGELU, nullptr, T_, HID_, C_, 2);
  // 10) FC2 (f32 out)
  gemm_bf16_kernel<<<dim3(C_ / 64, T_ / 128), 256, 0, stream>>>(
      (const bf16*)(ws + oGELU), fc2_w, fc2_b, ws + oMLP, nullptr, T_, C_, HID_, 0);
  // 11) z = ys + mlp + cross a2 -> d_out (ys stream slices as the 5 sources)
  const float* ysp = (const float*)(ws + oYS);
  combine_kernel<<<(unsigned)((totalTC + 255) / 256), 256, 0, stream>>>(
      ysp + 0 * (size_t)B_ * N_ * C_, ysp + 1 * (size_t)B_ * N_ * C_,
      ysp + 2 * (size_t)B_ * N_ * C_, ysp + 3 * (size_t)B_ * N_ * C_,
      ysp + 4 * (size_t)B_ * N_ * C_, (const float*)(ws + oMLP),
      (const float*)(ws + oH2B), a2_uw, a2_ub, (float*)d_out);
}
